// GPPTansAttention_38843684225059
// MI455X (gfx1250) — matrix operand order repaired and checked by reading, not compiled, not run on hardware
//
#include <hip/hip_runtime.h>
#include <hip/hip_bf16.h>

typedef __attribute__((ext_vector_type(16))) _Float16 v16h;
typedef __attribute__((ext_vector_type(8)))  float    v8f;

#define NN        50000            // original nodes
#define M_ROWS    50001            // +cls row
#define MPAD      50016            // padded to multiple of 16
#define NTILES    3126             // MPAD/16
#define DIM       128
#define NEDGES    800000
#define NHEADS    8
#define HDIM      16

// ---------------------------------------------------------------------------
// Prep: h16[MPAD][128] = f16 of [0; x; 0-pad]
// ---------------------------------------------------------------------------
__global__ void gppt_prep_h(const float* __restrict__ x, _Float16* __restrict__ h16) {
    size_t i = (size_t)blockIdx.x * blockDim.x + threadIdx.x;
    size_t tot = (size_t)MPAD * DIM;
    if (i >= tot) return;
    int row = (int)(i >> 7);
    int col = (int)(i & 127);
    float v = 0.0f;
    if (row >= 1 && row <= NN) v = x[(size_t)(row - 1) * DIM + col];
    h16[i] = (_Float16)v;
}

// ---------------------------------------------------------------------------
// Prep: pack W directly into the WMMA B-operand register image.
// Wpack[sel][nt][kt][lane][j] (f16), j=0..15:
//   lane's 16 values are W[k][col] with k = kt*32 + (lane>=16?8:0) + (j>=8?16:0) + (j&7)
//   (table order, matching A) and col = nt*16 + (lane&15).
// GEMM then loads one aligned 32B v16h per (nt,kt) -> two global_load_b128.
// Also copies the three biases into biasf[3][128].
// ---------------------------------------------------------------------------
__global__ void gppt_pack_w(const float* __restrict__ Wq, const float* __restrict__ Wk,
                            const float* __restrict__ Wv, const float* __restrict__ bq,
                            const float* __restrict__ bk, const float* __restrict__ bv,
                            _Float16* __restrict__ Wpack, float* __restrict__ biasf) {
    int i = blockIdx.x * blockDim.x + threadIdx.x;
    if (i < 3 * DIM * DIM) {                       // 3*8*4*32*16 == 3*128*128
        const int j    = i & 15;
        const int lane = (i >> 4) & 31;
        const int kt   = (i >> 9) & 3;
        const int nt   = (i >> 11) & 7;
        const int sel  = i >> 14;
        const int k    = kt * 32 + ((lane >> 4) << 3) + ((j >> 3) << 4) + (j & 7);
        const int col  = nt * 16 + (lane & 15);
        const float* src = (sel == 0) ? Wq : (sel == 1) ? Wk : Wv;
        Wpack[i] = (_Float16)src[(size_t)k * DIM + col];
    }
    if (i < 3 * DIM) {
        const int w = i / DIM;
        const int j = i % DIM;
        const float* src = (w == 0) ? bq : (w == 1) ? bk : bv;
        biasf[i] = src[j];
    }
}

// ---------------------------------------------------------------------------
// Zero accumulators (wV + Z region, contiguous floats)
// ---------------------------------------------------------------------------
__global__ void gppt_zero(float* __restrict__ p, size_t n) {
    size_t i = (size_t)blockIdx.x * blockDim.x + threadIdx.x;
    if (i < n) p[i] = 0.0f;
}

// ---------------------------------------------------------------------------
// QKV GEMM via v_wmma_f32_16x16x32_f16.
// grid.y in {0,1,2} selects (Wq,bq)->Q, (Wk,bk)->K, (Wv,bv)->V.
// Block = 256 threads = 8 waves; each wave owns one 16-row tile and computes
// all 128 output columns: A (16x128 f16) in 4 v16h regs, B from the packed
// operand image (1 aligned 32B load per tile), 8 n-tiles x 4 k-steps
// = 32 WMMAs per wave.
// ---------------------------------------------------------------------------
__global__ void gppt_qkv_gemm(const _Float16* __restrict__ h16,
                              const _Float16* __restrict__ Wpack,
                              const float* __restrict__ biasf,
                              float* __restrict__ qkv) {
    const int lane = threadIdx.x & 31;
    const int wave = threadIdx.x >> 5;
    const int row_tile = blockIdx.x * 8 + wave;
    if (row_tile >= NTILES) return;               // wave-uniform exit (EXEC stays all-1)
    const int sel = blockIdx.y;                   // 0=Q 1=K 2=V
    const int row_base = row_tile * 16;

    float* __restrict__ out = qkv + (size_t)sel * MPAD * DIM;

    // ---- A operand: 16-bit A-matrix 16x32 lane layout (ISA 7.12.2) ----
    // lane 0-15: M=lane,     K in {0..7}  (VGPR0-3) and {16..23} (VGPR4-7)
    // lane16-31: M=lane-16,  K in {8..15}           and {24..31}
    const int m  = lane & 15;
    const int kb = (lane >> 4) * 8;
    const _Float16* __restrict__ arow = h16 + (size_t)(row_base + m) * DIM;
    v16h a[4];
#pragma unroll
    for (int kt = 0; kt < 4; ++kt) {
#pragma unroll
        for (int v = 0; v < 8; ++v) {
            const int k = kt * 32 + kb + ((v >> 2) << 4) + ((v & 3) << 1);
            a[kt][2 * v]     = arow[k];
            a[kt][2 * v + 1] = arow[k + 1];
        }
    }

    const int n = lane & 15;
    // per-lane base into the packed B image: [sel][nt][kt][lane][16]
    const _Float16* __restrict__ wp =
        Wpack + ((size_t)sel * 8 * 4 * 32 + (size_t)lane) * 16;

#pragma unroll
    for (int nt = 0; nt < 8; ++nt) {
        const int col = nt * 16 + n;
        const float b = biasf[sel * DIM + col];
        v8f c;
#pragma unroll
        for (int r = 0; r < 8; ++r) c[r] = b;      // C init = bias (broadcast down rows)

#pragma unroll
        for (int kt = 0; kt < 4; ++kt) {
            const v16h bm = *(const v16h*)(wp + ((size_t)nt * 4 + kt) * 32 * 16);
            c = __builtin_amdgcn_wmma_f32_16x16x32_f16(
                    false, a[kt], false, bm, (short)0, c, false, false);
        }

        // ---- D store: 32-bit C/D 16x16 layout: VGPR r -> M = r + (lane<16?0:8)
        const int mhi = (lane >> 4) * 8;
#pragma unroll
        for (int r = 0; r < 8; ++r) {
            const int row = row_base + mhi + r;    // pad rows exist in qkv buffer
            out[(size_t)row * DIM + col] = c[r];
        }
    }
}

// ---------------------------------------------------------------------------
// Edge attention: 128 threads per edge (thread = head*16 + dim), 16-lane
// shuffle reduction for the per-head dot, atomic scatter-add into wV and Z.
// Edges: e < NN -> cls edge (src=0, dst=e+1); else graph edge (+1 shift).
// ---------------------------------------------------------------------------
__global__ void gppt_edge(const float* __restrict__ Q, const float* __restrict__ K,
                          const float* __restrict__ V, const int* __restrict__ ei,
                          float* __restrict__ wV, float* __restrict__ Z) {
    const int t     = threadIdx.x & 127;          // (head, dim) within edge
    const int grp   = threadIdx.x >> 7;           // edge slot within block
    const int gperb = blockDim.x >> 7;
    const int head  = t >> 4;
    const int dim   = t & 15;
    const long long total = (long long)NN + NEDGES;

    for (long long e = (long long)blockIdx.x * gperb + grp; e < total;
         e += (long long)gridDim.x * gperb) {
        int src, dst;
        if (e < NN) {
            src = 0;
            dst = (int)e + 1;
        } else {
            const int ee = (int)(e - NN);
            src = ei[ee] + 1;
            dst = ei[NEDGES + ee] + 1;
        }
        const float* __restrict__ vrow = V + (size_t)src * DIM;
        __builtin_prefetch(vrow + t, 0, 0);       // global_prefetch_b8 for the V gather

        float p = K[(size_t)src * DIM + t] * Q[(size_t)dst * DIM + t];
        // reduce the 16 lanes of this head (two heads per wave32)
        p += __shfl_xor(p, 8, 16);
        p += __shfl_xor(p, 4, 16);
        p += __shfl_xor(p, 2, 16);
        p += __shfl_xor(p, 1, 16);
        float s = p * 0.25f;                      // 1/sqrt(HDIM=16)
        s = fminf(5.0f, fmaxf(-5.0f, s));
        const float score = __expf(s);

        atomicAdd(&wV[(size_t)dst * DIM + t], vrow[t] * score);
        if (dim == 0) atomicAdd(&Z[(size_t)dst * NHEADS + head], score);
    }
}

// ---------------------------------------------------------------------------
// out[i-1][c] = wV[i][c] / (Z[i][head(c)] + 1e-6), i = 1..NN
// ---------------------------------------------------------------------------
__global__ void gppt_norm(const float* __restrict__ wV, const float* __restrict__ Z,
                          float* __restrict__ out) {
    size_t i = (size_t)blockIdx.x * blockDim.x + threadIdx.x;
    const size_t tot = (size_t)NN * DIM;
    if (i >= tot) return;
    const int node = (int)(i >> 7) + 1;
    const int c    = (int)(i & 127);
    const float z  = Z[(size_t)node * NHEADS + (c >> 4)];
    out[i] = wV[(size_t)node * DIM + c] / (z + 1e-6f);
}

// ---------------------------------------------------------------------------
extern "C" void kernel_launch(void* const* d_in, const int* in_sizes, int n_in,
                              void* d_out, int out_size, void* d_ws, size_t ws_size,
                              hipStream_t stream) {
    const float* x  = (const float*)d_in[0];
    const int*   ei = (const int*)d_in[1];
    const float* Wq = (const float*)d_in[2];
    const float* bq = (const float*)d_in[3];
    const float* Wk = (const float*)d_in[4];
    const float* bk = (const float*)d_in[5];
    const float* Wv = (const float*)d_in[6];
    const float* bv = (const float*)d_in[7];
    float* out = (float*)d_out;

    // Workspace layout (256B-aligned sections)
    constexpr size_t OFF_H16   = 0;
    constexpr size_t SZ_H16    = (size_t)MPAD * DIM * sizeof(_Float16);   // 12.8 MB
    constexpr size_t OFF_WPACK = OFF_H16 + SZ_H16;
    constexpr size_t SZ_WPACK  = (size_t)3 * DIM * DIM * sizeof(_Float16);
    constexpr size_t OFF_BIAS  = OFF_WPACK + SZ_WPACK;
    constexpr size_t SZ_BIAS   = (size_t)3 * DIM * sizeof(float);
    constexpr size_t OFF_QKV   = (OFF_BIAS + SZ_BIAS + 255) & ~(size_t)255;
    constexpr size_t SZ_ONE    = (size_t)MPAD * DIM * sizeof(float);      // 25.6 MB each
    constexpr size_t OFF_WV    = OFF_QKV + 3 * SZ_ONE;
    constexpr size_t OFF_Z     = OFF_WV + SZ_ONE;                          // Z follows wV

    char* ws = (char*)d_ws;
    _Float16* h16   = (_Float16*)(ws + OFF_H16);
    _Float16* Wpack = (_Float16*)(ws + OFF_WPACK);
    float*    biasf = (float*)(ws + OFF_BIAS);
    float*    qkv   = (float*)(ws + OFF_QKV);
    float*    wV    = (float*)(ws + OFF_WV);
    float*    Z     = (float*)(ws + OFF_Z);

    // 1) build padded f16 h
    {
        const size_t tot = (size_t)MPAD * DIM;
        gppt_prep_h<<<dim3((unsigned)((tot + 255) / 256)), dim3(256), 0, stream>>>(x, h16);
    }
    // 2) pack weights into WMMA B-operand image + copy biases
    gppt_pack_w<<<dim3((3 * DIM * DIM + 255) / 256), dim3(256), 0, stream>>>(
        Wq, Wk, Wv, bq, bk, bv, Wpack, biasf);
    // 3) zero wV + Z (contiguous floats)
    {
        const size_t nz = (size_t)MPAD * DIM + (size_t)MPAD * NHEADS;
        gppt_zero<<<dim3((unsigned)((nz + 255) / 256)), dim3(256), 0, stream>>>(wV, nz);
    }
    // 4) WMMA QKV projection: grid.y selects Q/K/V
    gppt_qkv_gemm<<<dim3((NTILES + 7) / 8, 3), dim3(256), 0, stream>>>(h16, Wpack, biasf, qkv);
    // 5) edge-parallel attention scatter
    {
        const float* Qm = qkv;
        const float* Km = qkv + (size_t)MPAD * DIM;
        const float* Vm = qkv + (size_t)2 * MPAD * DIM;
        gppt_edge<<<dim3(8192), dim3(256), 0, stream>>>(Qm, Km, Vm, ei, wV, Z);
    }
    // 6) normalize and emit rows 1..NN
    {
        const size_t tot = (size_t)NN * DIM;
        gppt_norm<<<dim3((unsigned)((tot + 255) / 256)), dim3(256), 0, stream>>>(wV, Z, out);
    }
    (void)in_sizes; (void)n_in; (void)out_size; (void)ws_size;
}
